// GraphConvolution_24739011625684
// MI455X (gfx1250) — matrix operand order repaired and checked by reading, not compiled, not run on hardware
//
#include <hip/hip_runtime.h>

#define NROWS 8192
#define FEAT  256
#define KSTEP 32

typedef __attribute__((ext_vector_type(2)))  float    v2f;
typedef __attribute__((ext_vector_type(8)))  float    v8f;
typedef __attribute__((ext_vector_type(8)))  _Float16 v8h;
typedef __attribute__((ext_vector_type(16))) _Float16 v16h;
typedef int gv4i __attribute__((vector_size(16)));   // matches builtin param type

// ---- CDNA5 async global->LDS copy (ASYNCcnt-tracked), builtin or asm ------
__device__ __forceinline__ void async_load_b128(const _Float16* g, _Float16* l) {
#if __has_builtin(__builtin_amdgcn_global_load_async_to_lds_b128)
    __builtin_amdgcn_global_load_async_to_lds_b128(
        (gv4i*)g, (gv4i*)l, 0, 0);
#else
    asm volatile("global_load_async_to_lds_b128 %0, %1, off"
                 :: "v"((unsigned)(size_t)l), "v"(g) : "memory");
#endif
}

__device__ __forceinline__ void wait_async0() {
#if __has_builtin(__builtin_amdgcn_s_wait_asynccnt)
    __builtin_amdgcn_s_wait_asynccnt(0);
#else
    asm volatile("s_wait_asynccnt 0x0" ::: "memory");
#endif
}

// ---------------------------------------------------------------------------
// Phase 1: Ht[k][j][i] = (f16) sum_c V[i][c] * w_k[c][j]
// f32 WMMA (16x16x4), exact f32 math; only the final store rounds to f16.
// Block: 256 thr = 8 waves = 2 mtiles x 4 nslices. Grid: 8192/32 = 256.
// ---------------------------------------------------------------------------
__global__ __launch_bounds__(256)
void gcn_phase1(const float* __restrict__ V,
                const float* __restrict__ w1,
                const float* __restrict__ w2,
                const float* __restrict__ w3,
                _Float16* __restrict__ Ht)      // [3][FEAT][NROWS]
{
    const int lane = threadIdx.x & 31;
    const int wave = threadIdx.x >> 5;
    const int mt   = wave >> 2;                 // 0..1
    const int ns   = wave & 3;                  // 0..3  (64 cols each)
    const int row0 = blockIdx.x * 32 + mt * 16;
    const int lrow = lane & 15;
    const int hi   = lane >> 4;                 // 0 or 1

    const float* W[3] = { w1, w2, w3 };

    v8f acc[3][4] = {};

    // A (16x4 f32): lane row = lrow, K = kk + hi*2 + {0,1}  -> contiguous 8B
    const float* aptr = V + (size_t)(row0 + lrow) * FEAT + hi * 2;

    for (int kk = 0; kk < FEAT; kk += 4) {
        v2f a = *(const v2f*)(aptr + kk);
#pragma unroll
        for (int k = 0; k < 3; ++k) {
            const float* wp = W[k] + (size_t)(kk + hi * 2) * FEAT + (ns * 64 + lrow);
#pragma unroll
            for (int nt = 0; nt < 4; ++nt) {
                v2f b;
                b.x = wp[nt * 16];
                b.y = wp[nt * 16 + FEAT];
                acc[k][nt] = __builtin_amdgcn_wmma_f32_16x16x4_f32(
                    false, a, false, b, (short)0, acc[k][nt], false, false);
            }
        }
    }

    // C/D: lane<16 -> (M=v, N=lane); lane>=16 -> (M=v+8, N=lane-16)
#pragma unroll
    for (int k = 0; k < 3; ++k)
#pragma unroll
        for (int nt = 0; nt < 4; ++nt) {
            v8h h;
#pragma unroll
            for (int v = 0; v < 8; ++v) h[v] = (_Float16)acc[k][nt][v];
            _Float16* dst = Ht + ((size_t)k * FEAT + (ns * 64 + nt * 16 + lrow)) * NROWS
                               + row0 + hi * 8;
            *(v8h*)dst = h;
        }
}

// ---------------------------------------------------------------------------
// Phase 2: out[i][j] = sum_k sum_p (adj[i][p]==k) * H_k[p][j] + bias[j]
// Single streaming pass over adj (256 MB -> HBM bound). Double-buffered LDS:
// B tiles filled by GLOBAL_LOAD_ASYNC_TO_LDS_B128, masks decoded once per adj
// element and stored in contiguous K order (position p holds K p; half h reads K 16h+s)
// so every A fragment is one contiguous 32B LDS read. 12 fused f16 WMMAs per
// K-step (3 edge types x 4 N-tiles) into one f32 accumulator.
// Block: 256 thr = 8 waves = 2 mtiles x 4 nslices; tile 32 rows x 256 cols.
// ---------------------------------------------------------------------------
__global__ __launch_bounds__(256)
void gcn_phase2(const int* __restrict__ adj,
                const _Float16* __restrict__ Ht,   // [3][FEAT][NROWS]
                const float* __restrict__ bias,
                float* __restrict__ out)
{
    __shared__ __attribute__((aligned(32))) _Float16 smask[2][3][2][16][KSTEP]; // 12 KB
    __shared__ __attribute__((aligned(32))) _Float16 sB[2][3][FEAT][KSTEP];     // 96 KB
    const int BUFH = 3 * FEAT * KSTEP;   // halves per sB buffer

    const int t    = threadIdx.x;
    const int lane = t & 31;
    const int wave = t >> 5;
    const int mt   = wave >> 2;
    const int ns   = wave & 3;
    const int row0 = blockIdx.x * 32;
    const int lrow = lane & 15;
    const int hi   = lane >> 4;

    // ---- adj/mask fill mapping: thread -> row (t>>3), 4-col group ((t&7)*4)
    const int frow = t >> 3;
    const int fmt  = frow >> 4, fr = frow & 15;
    const int pmap[4] = {0, 8, 16, 24};            // K-group -> LDS position (identity)
    const int pos  = pmap[(t & 7) >> 1] + (t & 1) * 4;
    const int* adj_p = adj + (size_t)(row0 + frow) * NROWS + (t & 7) * 4;

    // ---- async B fill mapping: thread covers 16B chunk (t&3) of row (t>>2),
    //      12 chunks stride 64 rows: sB[buf] flat row rr=k*256+n of 32 halves.
    const _Float16* gB = Ht + (size_t)(t >> 2) * NROWS + (t & 3) * 8;
    _Float16*       lB = &sB[0][0][0][0] + (t >> 2) * KSTEP + (t & 3) * 8;

    auto store_masks = [&](int b, int4 v) {
#pragma unroll
        for (int k = 0; k < 3; ++k) {
            union { _Float16 h[4]; unsigned long long u; } p;
            p.h[0] = (v.x == k + 1) ? (_Float16)1.0f : (_Float16)0.0f;
            p.h[1] = (v.y == k + 1) ? (_Float16)1.0f : (_Float16)0.0f;
            p.h[2] = (v.z == k + 1) ? (_Float16)1.0f : (_Float16)0.0f;
            p.h[3] = (v.w == k + 1) ? (_Float16)1.0f : (_Float16)0.0f;
            *(unsigned long long*)&smask[b][k][fmt][fr][pos] = p.u;
        }
    };

    v8f acc[4] = {};                               // fused over all 3 edge types

    // ---- prologue: fill buffer 0 ----
#pragma unroll
    for (int c = 0; c < 12; ++c)
        async_load_b128(gB + (size_t)c * 64 * NROWS, lB + c * 64 * KSTEP);
    store_masks(0, *(const int4*)adj_p);
    wait_async0();
    __syncthreads();

    int buf = 0;
    for (int kk = 0; kk < NROWS; kk += KSTEP) {
        const int  nxt  = buf ^ 1;
        const bool more = (kk + KSTEP) < NROWS;
        int4 avn = {};

        if (more) {
            // issue next tile's async B fill + adj load before computing
#pragma unroll
            for (int c = 0; c < 12; ++c)
                async_load_b128(gB + (size_t)c * 64 * NROWS + (kk + KSTEP),
                                lB + nxt * BUFH + c * 64 * KSTEP);
            avn = *(const int4*)(adj_p + kk + KSTEP);
            if (kk + 8 * KSTEP < NROWS)
                __builtin_prefetch(adj_p + kk + 8 * KSTEP, 0, 0);  // global_prefetch_b8
        }

        // ---- compute on current buffer ----
        v16h A[3];
#pragma unroll
        for (int k = 0; k < 3; ++k)   // contiguous K layout -> one 32B read per lane
            A[k] = *(const v16h*)&smask[buf][k][mt][lrow][hi * 16];

#pragma unroll
        for (int nt = 0; nt < 4; ++nt) {
            const int n = ns * 64 + nt * 16 + lrow;   // B: lane col, K = hi*16+e
#pragma unroll
            for (int k = 0; k < 3; ++k) {
                v16h B = *(const v16h*)&sB[buf][k][n][hi * 16];
                acc[nt] = __builtin_amdgcn_wmma_f32_16x16x32_f16(
                    false, A[k], false, B, (short)0, acc[nt], false, false);
            }
        }

        if (more) store_masks(nxt, avn);   // after compute: old masks consumed
        wait_async0();
        __syncthreads();
        buf = nxt;
    }

    // ---- epilogue: bias + f32 store ----
    const int orow = row0 + mt * 16 + hi * 8;
#pragma unroll
    for (int nt = 0; nt < 4; ++nt) {
        const int n = ns * 64 + nt * 16 + lrow;
        const float b = bias[n];
        float* op = out + (size_t)orow * FEAT + n;
#pragma unroll
        for (int v = 0; v < 8; ++v)
            op[(size_t)v * FEAT] = acc[nt][v] + b;
    }
}

// ---------------------------------------------------------------------------
extern "C" void kernel_launch(void* const* d_in, const int* in_sizes, int n_in,
                              void* d_out, int out_size, void* d_ws, size_t ws_size,
                              hipStream_t stream) {
    const float* V    = (const float*)d_in[0];
    const int*   adj  = (const int*)  d_in[1];
    const float* w1   = (const float*)d_in[2];
    const float* w2   = (const float*)d_in[3];
    const float* w3   = (const float*)d_in[4];
    const float* bias = (const float*)d_in[5];
    float*       out  = (float*)d_out;

    _Float16* Ht = (_Float16*)d_ws;   // needs 3*256*8192*2 = 12 MB

    gcn_phase1<<<dim3(NROWS / 32), dim3(256), 0, stream>>>(V, w1, w2, w3, Ht);
    gcn_phase2<<<dim3(NROWS / 32), dim3(256), 0, stream>>>(adj, Ht, bias, out);
}
